// PointNet2_4080218931823
// MI455X (gfx1250) — matrix operand order repaired and checked by reading, not compiled, not run on hardware
//
#include <hip/hip_runtime.h>
#include <hip/hip_bf16.h>
#include <math.h>

typedef __attribute__((ext_vector_type(16))) _Float16 v16h;
typedef __attribute__((ext_vector_type(8)))  _Float16 v8h;
typedef __attribute__((ext_vector_type(8)))  float    v8f;

#define BN_EPS 1e-5f

// ---------------------------------------------------------------------------
// Farthest point sampling: one block per batch, LDS argmax reduction per step.
// ---------------------------------------------------------------------------
template<int MAXPER>
__global__ void fps_kernel(const float* __restrict__ xyz, int* __restrict__ out,
                           int N, int npoint) {
    const int b = blockIdx.x;
    const int t = threadIdx.x;
    const int T = blockDim.x;
    const float* x = xyz + (size_t)b * N * 3;
    int* o = out + (size_t)b * npoint;

    float dist[MAXPER];
#pragma unroll
    for (int i = 0; i < MAXPER; ++i) dist[i] = 1e30f;

    __shared__ float sval[1024];
    __shared__ int   sidx[1024];

    int far = 0;
    for (int it = 0; it < npoint; ++it) {
        if (t == 0) o[it] = far;
        const float cx = x[far*3+0], cy = x[far*3+1], cz = x[far*3+2];
        float best = -1.0f; int bi = 0;
#pragma unroll
        for (int i = 0; i < MAXPER; ++i) {
            int j = i * T + t;
            if (j < N) {
                float dx = x[j*3+0]-cx, dy = x[j*3+1]-cy, dz = x[j*3+2]-cz;
                float d  = dx*dx + dy*dy + dz*dz;
                float nd = fminf(dist[i], d);
                dist[i] = nd;
                if (nd > best) { best = nd; bi = j; }
            }
        }
        sval[t] = best; sidx[t] = bi;
        __syncthreads();
        for (int s = T >> 1; s > 0; s >>= 1) {
            if (t < s && sval[t+s] > sval[t]) { sval[t] = sval[t+s]; sidx[t] = sidx[t+s]; }
            __syncthreads();
        }
        far = sidx[0];
        __syncthreads();
    }
}

__global__ void gather_xyz_kernel(const float* __restrict__ xyz, const int* __restrict__ fidx,
                                  float* __restrict__ out, int B, int N, int M) {
    int gid = blockIdx.x * blockDim.x + threadIdx.x;
    if (gid >= B * M) return;
    int b = gid / M;
    int j = fidx[gid];
    const float* s = xyz + ((size_t)b * N + j) * 3;
    float* d = out + (size_t)gid * 3;
    d[0] = s[0]; d[1] = s[1]; d[2] = s[2];
}

// Ball query: first nsample indices (ascending) within radius, pad with first.
__global__ void ball_query_kernel(const float* __restrict__ xyz, const float* __restrict__ nxyz,
                                  int* __restrict__ idx, int B, int N, int M, int ns, float r2) {
    int gid = blockIdx.x * blockDim.x + threadIdx.x;
    if (gid >= B * M) return;
    int b = gid / M;
    const float* x = xyz + (size_t)b * N * 3;
    const float* c = nxyz + (size_t)gid * 3;
    float cx = c[0], cy = c[1], cz = c[2];
    int* o = idx + (size_t)gid * ns;
    int cnt = 0, first = 0;
    for (int j = 0; j < N && cnt < ns; ++j) {
        float dx = x[j*3+0]-cx, dy = x[j*3+1]-cy, dz = x[j*3+2]-cz;
        float d = dx*dx + dy*dy + dz*dz;
        if (d <= r2) { if (cnt == 0) first = j; o[cnt++] = j; }
    }
    for (; cnt < ns; ++cnt) o[cnt] = first;
}

// SA1 grouping: rows = centered xyz (3ch), zero-padded to Kp.
__global__ void group1_kernel(const float* __restrict__ xyz, const float* __restrict__ nxyz,
                              const int* __restrict__ idx, _Float16* __restrict__ act,
                              int B, int N, int M, int ns, int Kp) {
    int gid = blockIdx.x * blockDim.x + threadIdx.x;
    if (gid >= B * M * ns) return;
    int bm = gid / ns;
    int b = bm / M;
    int j = idx[gid];
    const float* p = xyz + ((size_t)b * N + j) * 3;
    const float* c = nxyz + (size_t)bm * 3;
    _Float16* a = act + (size_t)gid * Kp;
    a[0] = (_Float16)(p[0]-c[0]); a[1] = (_Float16)(p[1]-c[1]); a[2] = (_Float16)(p[2]-c[2]);
    for (int k = 3; k < Kp; ++k) a[k] = (_Float16)0.f;
}

// SA2 grouping: [centered xyz(3), gathered features(C)], zero-padded to Kp.
__global__ void group2_kernel(const float* __restrict__ xyz, const float* __restrict__ nxyz,
                              const int* __restrict__ idx, const _Float16* __restrict__ feats,
                              int C, _Float16* __restrict__ act,
                              int B, int N, int M, int ns, int Kp) {
    int gid = blockIdx.x * blockDim.x + threadIdx.x;
    if (gid >= B * M * ns) return;
    int bm = gid / ns;
    int b = bm / M;
    int j = idx[gid];
    const float* p = xyz + ((size_t)b * N + j) * 3;
    const float* c = nxyz + (size_t)bm * 3;
    _Float16* a = act + (size_t)gid * Kp;
    a[0] = (_Float16)(p[0]-c[0]); a[1] = (_Float16)(p[1]-c[1]); a[2] = (_Float16)(p[2]-c[2]);
    const _Float16* f = feats + ((size_t)b * N + j) * C;
    for (int cc = 0; cc < C; ++cc) a[3+cc] = f[cc];
    for (int k = 3 + C; k < Kp; ++k) a[k] = (_Float16)0.f;
}

__global__ void maxpool_kernel(const _Float16* __restrict__ in, _Float16* __restrict__ out,
                               int BM, int ns, int C) {
    int gid = blockIdx.x * blockDim.x + threadIdx.x;
    if (gid >= BM * C) return;
    int c = gid % C, bm = gid / C;
    const _Float16* p = in + ((size_t)bm * ns) * C + c;
    float m = (float)p[0];
    for (int s = 1; s < ns; ++s) m = fmaxf(m, (float)p[(size_t)s * C]);
    out[(size_t)bm * C + c] = (_Float16)m;
}

// 3-NN inverse-distance interpolation + concat into padded f16 activation rows.
__global__ void fp_interp_kernel(const float* __restrict__ xyz1, const float* __restrict__ xyz2,
                                 const _Float16* __restrict__ pts1h, int c1h,
                                 const float* __restrict__ pts1f, int c1f,
                                 const _Float16* __restrict__ pts2, int c2,
                                 _Float16* __restrict__ act, int Kp,
                                 int B, int N1, int N2) {
    int gid = blockIdx.x * blockDim.x + threadIdx.x;
    if (gid >= B * N1) return;
    int b = gid / N1;
    const float* q = xyz1 + (size_t)gid * 3;
    const float* x2 = xyz2 + (size_t)b * N2 * 3;
    float qx = q[0], qy = q[1], qz = q[2];
    float bd0 = 1e30f, bd1 = 1e30f, bd2 = 1e30f; int i0 = 0, i1 = 0, i2 = 0;
    for (int j = 0; j < N2; ++j) {
        float dx = x2[j*3+0]-qx, dy = x2[j*3+1]-qy, dz = x2[j*3+2]-qz;
        float d = dx*dx + dy*dy + dz*dz;
        if (d < bd0)      { bd2=bd1; i2=i1; bd1=bd0; i1=i0; bd0=d; i0=j; }
        else if (d < bd1) { bd2=bd1; i2=i1; bd1=d; i1=j; }
        else if (d < bd2) { bd2=d; i2=j; }
    }
    float w0 = 1.f/(sqrtf(fmaxf(bd0,0.f)) + 1e-8f);
    float w1 = 1.f/(sqrtf(fmaxf(bd1,0.f)) + 1e-8f);
    float w2 = 1.f/(sqrtf(fmaxf(bd2,0.f)) + 1e-8f);
    float wsum = w0 + w1 + w2; w0 /= wsum; w1 /= wsum; w2 /= wsum;
    _Float16* a = act + (size_t)gid * Kp;
    int k = 0;
    if (pts1h) { const _Float16* f = pts1h + (size_t)gid * c1h; for (int c = 0; c < c1h; ++c) a[k++] = f[c]; }
    if (pts1f) { const float* f = pts1f + (size_t)gid * c1f;    for (int c = 0; c < c1f; ++c) a[k++] = (_Float16)f[c]; }
    const _Float16* f0 = pts2 + ((size_t)b * N2 + i0) * c2;
    const _Float16* f1 = pts2 + ((size_t)b * N2 + i1) * c2;
    const _Float16* f2 = pts2 + ((size_t)b * N2 + i2) * c2;
    for (int c = 0; c < c2; ++c)
        a[k++] = (_Float16)(w0*(float)f0[c] + w1*(float)f1[c] + w2*(float)f2[c]);
    for (; k < Kp; ++k) a[k] = (_Float16)0.f;
}

// Fold BN into weights: Wh[o,k] = W[o,k]*g/sqrt(rv+eps) (K zero-padded to 32),
// bias2[o] = (b-rm)*scale + be.
__global__ void prep_weights_kernel(const float* __restrict__ W, const float* __restrict__ b,
                                    const float* __restrict__ g, const float* __restrict__ be,
                                    const float* __restrict__ rm, const float* __restrict__ rv,
                                    _Float16* __restrict__ Wh, float* __restrict__ bias2,
                                    int cout, int cin, int Kp) {
    int gid = blockIdx.x * blockDim.x + threadIdx.x;
    if (gid >= cout * Kp) return;
    int o = gid / Kp, k = gid - o * Kp;
    float s = g[o] * rsqrtf(rv[o] + BN_EPS);
    Wh[gid] = (k < cin) ? (_Float16)(W[o*cin + k] * s) : (_Float16)0.f;
    if (k == 0) bias2[o] = (b[o] - rm[o]) * s + be[o];
}

// ---------------------------------------------------------------------------
// WMMA GEMM with 64x64 register tiling per wave32:
//   Out[M,ldo] = relu(f16 A[M,K] x f16 W[N,K]^T + bias), f32 accumulation.
// Per K-step: 4 A fragments (rows +0/16/32/48), 4 B fragments (cols
// +0/16/32/48), 16 v_wmma_f32_16x16x32_f16 -> 32 FLOP/byte of loaded data,
// 16 independent accumulation chains to cover WMMA latency.
// M%64==0, N%64==0, K%32==0 -> exact grid, EXEC all ones (WMMA requirement).
// ---------------------------------------------------------------------------
__global__ void __launch_bounds__(32)
wmma_gemm_bias_relu(const _Float16* __restrict__ A,
                    const _Float16* __restrict__ W,
                    const float* __restrict__ bias,
                    _Float16* __restrict__ Out,
                    int M, int N, int K, int ldo) {
    const int lane = threadIdx.x;
    const int lo16 = lane & 15;
    const int hi   = lane >> 4;
    const int m0 = blockIdx.x << 6;
    const int n0 = blockIdx.y << 6;

    const _Float16* aBase = A + (size_t)(m0 + lo16) * K + hi * 16;
    const _Float16* wBase = W + (size_t)(n0 + lo16) * K + hi * 16;
    const size_t rs = (size_t)16 * K;  // 16-row stride in halves

    v8f acc[4][4];
#pragma unroll
    for (int mi = 0; mi < 4; ++mi)
#pragma unroll
        for (int ni = 0; ni < 4; ++ni) acc[mi][ni] = (v8f){};

    for (int k0 = 0; k0 < K; k0 += 32) {
        v16h af[4], bf[4];
#pragma unroll
        for (int i = 0; i < 4; ++i) {
            v8h a0 = *reinterpret_cast<const v8h*>(aBase + i * rs + k0);
            v8h a1 = *reinterpret_cast<const v8h*>(aBase + i * rs + k0 + 8);
#pragma unroll
            for (int e = 0; e < 8; ++e) { af[i][e] = a0[e]; af[i][8+e] = a1[e]; }
            bf[i] = *reinterpret_cast<const v16h*>(wBase + i * rs + k0);
        }
#pragma unroll
        for (int mi = 0; mi < 4; ++mi)
#pragma unroll
            for (int ni = 0; ni < 4; ++ni)
                acc[mi][ni] = __builtin_amdgcn_wmma_f32_16x16x32_f16(
                    false, af[mi], false, bf[ni], (short)0, acc[mi][ni], false, false);
    }

#pragma unroll
    for (int ni = 0; ni < 4; ++ni) {
        const int col = n0 + (ni << 4) + lo16;
        const float bv = bias[col];
#pragma unroll
        for (int mi = 0; mi < 4; ++mi) {
            const int rbase = m0 + (mi << 4) + (hi << 3);
#pragma unroll
            for (int v = 0; v < 8; ++v) {
                float val = fmaxf(acc[mi][ni][v] + bv, 0.0f);
                Out[(size_t)(rbase + v) * ldo + col] = (_Float16)val;
            }
        }
    }
}

// Final 128->13 linear + log_softmax, scalar f32.
__global__ void logits_kernel(const _Float16* __restrict__ act, const float* __restrict__ Wc,
                              const float* __restrict__ bc, float* __restrict__ out, int BN) {
    int gid = blockIdx.x * blockDim.x + threadIdx.x;
    if (gid >= BN) return;
    const _Float16* a = act + (size_t)gid * 128;
    float lg[13];
    float mx = -1e30f;
    for (int o = 0; o < 13; ++o) {
        const float* w = Wc + o * 128;
        float s = bc[o];
        for (int c = 0; c < 128; ++c) s += (float)a[c] * w[c];
        lg[o] = s; mx = fmaxf(mx, s);
    }
    float se = 0.f;
    for (int o = 0; o < 13; ++o) se += __expf(lg[o] - mx);
    float lse = mx + __logf(se);
    float* op = out + (size_t)gid * 13;
    for (int o = 0; o < 13; ++o) op[o] = lg[o] - lse;
}

// ---------------------------------------------------------------------------
extern "C" void kernel_launch(void* const* d_in, const int* in_sizes, int n_in,
                              void* d_out, int out_size, void* d_ws, size_t ws_size,
                              hipStream_t stream) {
    (void)in_sizes; (void)n_in; (void)out_size; (void)ws_size;
    const int B = 4, N0 = 16384, M1 = 1024, M2 = 256, NS = 32;
    const float* xyz = (const float*)d_in[0];

    // Input flattening order: xyz, then 6 arrays per conv layer, then (Wc, bc).
    struct LDef { int base, cin, cout; };
    const LDef L[11] = {
        { 1,   3,  64}, { 7,  64,  64}, {13,  64, 128},   // sa1
        {19, 131, 128}, {25, 128, 128}, {31, 128, 256},   // sa2
        {37, 384, 256}, {43, 256, 128},                   // fp1
        {49, 131, 128}, {55, 128, 128},                   // fp0
        {61, 128, 128}                                    // cls conv
    };
    auto kp32 = [](int c) { return (c + 31) & ~31; };

    char* ws = (char*)d_ws;
    size_t off = 0;
    auto alloc = [&](size_t bytes) -> char* {
        char* p = ws + off;
        off = (off + bytes + 255) & ~(size_t)255;
        return p;
    };

    int*      fidx1 = (int*)alloc(sizeof(int) * B * M1);
    int*      fidx2 = (int*)alloc(sizeof(int) * B * M2);
    float*    l1x   = (float*)alloc(sizeof(float) * B * M1 * 3);
    float*    l2x   = (float*)alloc(sizeof(float) * B * M2 * 3);
    int*      bidx1 = (int*)alloc(sizeof(int) * B * M1 * NS);
    int*      bidx2 = (int*)alloc(sizeof(int) * B * M2 * NS);
    _Float16* p1    = (_Float16*)alloc(sizeof(_Float16) * (size_t)B * M1 * 128);
    _Float16* p2    = (_Float16*)alloc(sizeof(_Float16) * (size_t)B * M2 * 256);
    _Float16* p1n   = (_Float16*)alloc(sizeof(_Float16) * (size_t)B * M1 * 128);
    const size_t bufElems = (size_t)B * M1 * NS * 128;  // 131072 x 128, largest layer
    _Float16* bufA  = (_Float16*)alloc(sizeof(_Float16) * bufElems);
    _Float16* bufB  = (_Float16*)alloc(sizeof(_Float16) * bufElems);

    _Float16* Wh[11]; float* Bh[11];
    for (int i = 0; i < 11; ++i) {
        int Kp = kp32(L[i].cin);
        Wh[i] = (_Float16*)alloc(sizeof(_Float16) * (size_t)L[i].cout * Kp);
        Bh[i] = (float*)alloc(sizeof(float) * L[i].cout);
    }

    // BN-folded f16 weight prep (weights stay L2-resident afterwards).
    for (int i = 0; i < 11; ++i) {
        int Kp = kp32(L[i].cin);
        int total = L[i].cout * Kp;
        prep_weights_kernel<<<(total + 255) / 256, 256, 0, stream>>>(
            (const float*)d_in[L[i].base + 0], (const float*)d_in[L[i].base + 1],
            (const float*)d_in[L[i].base + 2], (const float*)d_in[L[i].base + 3],
            (const float*)d_in[L[i].base + 4], (const float*)d_in[L[i].base + 5],
            Wh[i], Bh[i], L[i].cout, L[i].cin, Kp);
    }

    auto gemm = [&](const _Float16* Ain, int li, _Float16* Oout, int M) {
        int N = L[li].cout, K = kp32(L[li].cin);
        dim3 g(M / 64, N / 64);
        wmma_gemm_bias_relu<<<g, 32, 0, stream>>>(Ain, Wh[li], Bh[li], Oout, M, N, K, N);
    };

    // ---- SA1: 16384 -> 1024 points, r=0.1, MLP [64,64,128] ----
    fps_kernel<16><<<B, 1024, 0, stream>>>(xyz, fidx1, N0, M1);
    gather_xyz_kernel<<<(B*M1 + 255)/256, 256, 0, stream>>>(xyz, fidx1, l1x, B, N0, M1);
    ball_query_kernel<<<(B*M1 + 127)/128, 128, 0, stream>>>(xyz, l1x, bidx1, B, N0, M1, NS, 0.01f);
    {
        int rows = B * M1 * NS;  // 131072
        group1_kernel<<<(rows + 255)/256, 256, 0, stream>>>(xyz, l1x, bidx1, bufA, B, N0, M1, NS, 32);
        gemm(bufA, 0, bufB, rows);
        gemm(bufB, 1, bufA, rows);
        gemm(bufA, 2, bufB, rows);
        maxpool_kernel<<<((B*M1*128) + 255)/256, 256, 0, stream>>>(bufB, p1, B*M1, NS, 128);
    }

    // ---- SA2: 1024 -> 256 points, r=0.2, MLP [128,128,256] ----
    fps_kernel<4><<<B, 256, 0, stream>>>(l1x, fidx2, M1, M2);
    gather_xyz_kernel<<<(B*M2 + 255)/256, 256, 0, stream>>>(l1x, fidx2, l2x, B, M1, M2);
    ball_query_kernel<<<(B*M2 + 127)/128, 128, 0, stream>>>(l1x, l2x, bidx2, B, M1, M2, NS, 0.04f);
    {
        int rows = B * M2 * NS;  // 32768
        group2_kernel<<<(rows + 255)/256, 256, 0, stream>>>(l1x, l2x, bidx2, p1, 128, bufA, B, M1, M2, NS, 160);
        gemm(bufA, 3, bufB, rows);
        gemm(bufB, 4, bufA, rows);
        gemm(bufA, 5, bufB, rows);
        maxpool_kernel<<<((B*M2*256) + 255)/256, 256, 0, stream>>>(bufB, p2, B*M2, NS, 256);
    }

    // ---- FP1: interpolate l2 features onto l1, concat p1 -> MLP [256,128] ----
    {
        int rows = B * M1;  // 4096
        fp_interp_kernel<<<(rows + 127)/128, 128, 0, stream>>>(
            l1x, l2x, p1, 128, (const float*)nullptr, 0, p2, 256, bufA, 384, B, M1, M2);
        gemm(bufA, 6, bufB, rows);
        gemm(bufB, 7, p1n, rows);
    }

    // ---- FP0: interpolate l1 features onto xyz, concat xyz -> MLP [128,128] ----
    // ---- then cls conv + 128->13 linear + log_softmax ----
    {
        int rows = B * N0;  // 65536
        fp_interp_kernel<<<(rows + 127)/128, 128, 0, stream>>>(
            xyz, l1x, (const _Float16*)nullptr, 0, xyz, 3, p1n, 128, bufA, 160, B, N0, M1);
        gemm(bufA, 8, bufB, rows);
        gemm(bufB, 9, bufA, rows);
        gemm(bufA, 10, bufB, rows);
        logits_kernel<<<(rows + 127)/128, 128, 0, stream>>>(
            bufB, (const float*)d_in[67], (const float*)d_in[68], (float*)d_out, rows);
    }
}
